// StanceAttn_3289944949421
// MI455X (gfx1250) — matrix operand order repaired and checked by reading, not compiled, not run on hardware
//
#include <hip/hip_runtime.h>
#include <hip/hip_bf16.h>
#include <stdint.h>
#include <math.h>

#define E_   300
#define EP_  320        // E padded to multiple of 32 for WMMA K
#define H_   256
#define B_   16
#define S_   2048
#define G_   1024       // 4H gate width
#define D2_  512        // 2H bilstm width
#define KC_  (EP_ + H_) // 576 combined K (x | h)
#define KT_  (KC_ / 32) // 18 k-tiles

typedef __bf16 bf16;
typedef __attribute__((ext_vector_type(16))) __bf16 bf16x16;
typedef __attribute__((ext_vector_type(8)))  __bf16 bf16x8;
typedef __attribute__((ext_vector_type(8)))  float  f32x8;

static __device__ __forceinline__ f32x8 wmma_bf16(bf16x16 a, bf16x16 b, f32x8 c) {
    // D = A(16x32 bf16) * B(32x16 bf16) + C(16x16 f32)
    return __builtin_amdgcn_wmma_f32_16x16x32_bf16(false, a, false, b, (short)0, c,
                                                   false, false);
}

static __device__ __forceinline__ float sigf(float x) {
    return 1.0f / (1.0f + __expf(-x));       // branch-free: v_exp_f32 path
}

static __device__ __forceinline__ float tanh_fast(float x) {
    // branch-free tanh via exp: saturates to +/-1 for large |x|
    return 1.0f - 2.0f / (__expf(2.0f * x) + 1.0f);
}

// ---------------------------------------------------------------------------
// xbf[b][s][k] = bf16(x[b][s][k]) for k<E, 0 for padding
__global__ void k_convert_x(const float* __restrict__ x, bf16* __restrict__ xbf) {
    int idx = blockIdx.x * 256 + threadIdx.x;
    if (idx >= B_ * S_ * EP_) return;
    int k  = idx % EP_;
    int bs = idx / EP_;
    float v = (k < E_) ? x[(size_t)bs * E_ + k] : 0.0f;
    xbf[idx] = (bf16)v;
}

// ---------------------------------------------------------------------------
// Pack Wc = [Wih (K 0..319, zero-padded 300..319) | Whh (K 320..575)] into
// WMMA B-fragment order: dst[((jt*KT + kt)*32 + lane)*16 + e]
// B layout (32x16, bf16): K = (e&7) + 8*(lane>>4) + 16*(e>>3); N = lane&15
__global__ void k_pack_wc(const float* __restrict__ Wih, const float* __restrict__ Whh,
                          bf16* __restrict__ dst) {
    int idx = blockIdx.x * 256 + threadIdx.x;
    if (idx >= 64 * KT_ * 32 * 16) return;
    int e    = idx & 15;
    int l    = (idx >> 4) & 31;
    int rest = idx >> 9;
    int kt   = rest % KT_;
    int jt   = rest / KT_;
    int n    = jt * 16 + (l & 15);
    int Kg   = kt * 32 + (e & 7) + ((l >> 4) << 3) + ((e >> 3) << 4);
    float v;
    if (Kg < EP_) v = (Kg < E_) ? Wih[(size_t)n * E_ + Kg] : 0.0f;
    else          v = Whh[(size_t)n * H_ + (Kg - EP_)];
    dst[idx] = (bf16)v;
}

// hW^T fragments for the attention GEMM: B[k][n] = hW[n][k]
__global__ void k_pack_hw(const float* __restrict__ hW, bf16* __restrict__ dst) {
    int idx = blockIdx.x * 256 + threadIdx.x;
    if (idx >= 32 * 16 * 32 * 16) return;
    int e    = idx & 15;
    int l    = (idx >> 4) & 31;
    int rest = idx >> 9;
    int kt   = rest % 16;
    int jt   = rest / 16;
    int n    = jt * 16 + (l & 15);
    int Kg   = kt * 32 + (e & 7) + ((l >> 4) << 3) + ((e >> 3) << 4);
    dst[idx] = (bf16)hW[(size_t)n * D2_ + Kg];
}

// ---------------------------------------------------------------------------
// t_proj[b][n] = target[b] . tW[n] + tb[n]   (tiny: 4.9 MF)
__global__ void k_tproj(const float* __restrict__ tgt, const float* __restrict__ tW,
                        const float* __restrict__ tb, float* __restrict__ tp) {
    int idx = blockIdx.x * 256 + threadIdx.x;
    if (idx >= B_ * D2_) return;
    int b = idx / D2_, n = idx % D2_;
    const float* xr = tgt + (size_t)b * E_;
    const float* wr = tW  + (size_t)n * E_;
    float acc = tb[n];
    for (int e = 0; e < E_; ++e) acc += xr[e] * wr[e];
    tp[idx] = acc;
}

// ---------------------------------------------------------------------------
// One direction of the LSTM per block (blockIdx.x: 0=fwd, 1=bwd).
// 1024 threads = 32 waves; wave w owns gate n-tiles 2w, 2w+1.
// Per step: gates = [x_t | h] @ [Wih | Whh]^T + b  via 18 k-tile bf16 WMMAs,
// then pointwise cell update with c held in registers (4 cells/thread).
// NOTE: the kt loop must stay (mostly) rolled — if fully unrolled, the 36
// loop-invariant weight-fragment loads get hoisted out of the t-loop and
// spill to scratch. unroll_count(2) keeps addresses kt-dependent (no hoist)
// while giving the scheduler two WMMAs worth of load/compute overlap.
__global__ __launch_bounds__(1024)
void k_bilstm(const bf16* __restrict__ xbf,
              const bf16* __restrict__ WcF, const bf16* __restrict__ WcB,
              const float* __restrict__ bF, const float* __restrict__ bB,
              float* __restrict__ ht, bf16* __restrict__ htbf) {
    const int  dir     = blockIdx.x;
    const bf16* Wc     = dir ? WcB : WcF;
    const float* bias  = dir ? bB : bF;
    const int  dirBase = dir ? H_ : 0;

    __shared__ __align__(16) bf16 Alds[B_][KC_]; // [x_t | h] bf16, 18 KB
    __shared__ float gLds[B_][G_];               // gates f32, 64 KB

    const int tid  = threadIdx.x;
    const int lane = tid & 31;
    const int wv   = tid >> 5;
    const int row  = lane & 15;
    const int hi8  = (lane >> 4) << 3; // 0 or 8
    const int jt0  = wv * 2, jt1 = wv * 2 + 1;

    // per-wave fragment base pointers; per-kt stride = 32 lanes * 16 bf16 = 512
    const bf16* wp0 = Wc + ((size_t)jt0 * KT_ * 32 + lane) * 16;
    const bf16* wp1 = Wc + ((size_t)jt1 * KT_ * 32 + lane) * 16;

    // zero hidden-state slice of A
    for (int i = tid; i < B_ * H_; i += 1024)
        Alds[i / H_][EP_ + (i % H_)] = (bf16)0.0f;

    // each thread owns 4 consecutive cells (em, en..en+3) of c
    const int flat = tid * 4;
    const int em = flat / H_;
    const int en = flat % H_;
    float c0 = 0.f, c1 = 0.f, c2 = 0.f, c3 = 0.f;

    const float bv0 = bias[jt0 * 16 + row];
    const float bv1 = bias[jt1 * 16 + row];

    for (int t = 0; t < S_; ++t) {
        const int sidx = dir ? (S_ - 1 - t) : t;

        // stage x_t (bf16) into A rows [0..319]
        for (int i = tid; i < B_ * (EP_ / 2); i += 1024) {
            int m = i / (EP_ / 2);
            int d = i % (EP_ / 2);
            const uint32_t* src = (const uint32_t*)(xbf + ((size_t)m * S_ + sidx) * EP_);
            ((uint32_t*)&Alds[m][0])[d] = src[d];
        }
        // prefetch next step's x rows (global_prefetch_b8); 64B granularity
        if (t + 1 < S_) {
            const int nsidx = dir ? (S_ - 2 - t) : (t + 1);
            for (int i = tid; i < B_ * 10; i += 1024) {
                int m = i / 10, chunk = i % 10;
                __builtin_prefetch(xbf + ((size_t)m * S_ + nsidx) * EP_ + chunk * 32, 0, 3);
            }
        }
        __syncthreads(); // x_t staged + h from previous step visible

        // gates = A @ Wc^T  (M=16 = batch, N=1024, K=576)
        {
            f32x8 acc0 = {}; f32x8 acc1 = {};
#pragma clang loop unroll_count(2)
            for (int kt = 0; kt < KT_; ++kt) {
                int kb = kt * 32 + hi8;
                union { bf16x16 v; bf16x8 h[2]; } a;
                a.h[0] = *(const bf16x8*)&Alds[row][kb];
                a.h[1] = *(const bf16x8*)&Alds[row][kb + 16];
                bf16x16 b0 = *(const bf16x16*)(wp0 + (size_t)kt * 512);
                bf16x16 b1 = *(const bf16x16*)(wp1 + (size_t)kt * 512);
                acc0 = wmma_bf16(a.v, b0, acc0);
                acc1 = wmma_bf16(a.v, b1, acc1);
            }
#pragma unroll
            for (int r = 0; r < 8; ++r) {
                int m = r + hi8;
                gLds[m][jt0 * 16 + row] = acc0[r] + bv0;
                gLds[m][jt1 * 16 + row] = acc1[r] + bv1;
            }
        }
        __syncthreads(); // gates complete

        // pointwise LSTM cell: i,f,g,o -> c,h (branch-free transcendentals)
        {
            const float* gm = &gLds[em][0];
            float hv[4];
#pragma unroll
            for (int q = 0; q < 4; ++q) {
                int n = en + q;
                float ig = sigf(gm[n]);
                float fg = sigf(gm[n + H_]);
                float gg = tanh_fast(gm[n + 2 * H_]);
                float og = sigf(gm[n + 3 * H_]);
                float& c = (q == 0) ? c0 : (q == 1) ? c1 : (q == 2) ? c2 : c3;
                c = fg * c + ig * gg;
                float h = og * tanh_fast(c);
                hv[q] = h;
                Alds[em][EP_ + n] = (bf16)h; // next step's A
            }
            size_t o = ((size_t)em * S_ + sidx) * D2_ + dirBase + en;
#pragma unroll
            for (int q = 0; q < 4; ++q) { ht[o + q] = hv[q]; htbf[o + q] = (bf16)hv[q]; }
        }
        // no barrier needed here: next-iter x-staging touches disjoint LDS,
        // and the loop-top __syncthreads orders h-writes before the next GEMM.
    }
}

// ---------------------------------------------------------------------------
// Fused attention-energy GEMM + score reduction. Block = one 16-row (b,s) tile
// over all 512 columns; never materializes e.
// scores[b][s] = sum_n final[b][n] * tanh(tp[b][n] + (ht[b][s] . hW[n]))
__global__ __launch_bounds__(256)
void k_scores(const bf16* __restrict__ htbf, const bf16* __restrict__ hWf,
              const float* __restrict__ tp, const float* __restrict__ ht,
              float* __restrict__ scores) {
    const int mt  = blockIdx.x;           // 0 .. B*S/16-1
    const int b   = mt / (S_ / 16);
    const int s0  = (mt % (S_ / 16)) * 16;
    const int tid = threadIdx.x, lane = tid & 31, wv = tid >> 5;
    const int row = lane & 15, hi8 = (lane >> 4) << 3;

    __shared__ float sLds[16];
    if (tid < 16) sLds[tid] = 0.0f;
    __syncthreads();

    const size_t arow = ((size_t)b * S_ + s0 + row) * D2_;
    const int jtb = wv * 4;               // 8 waves x 4 n-tiles = 32 n-tiles
    f32x8 acc[4] = {};
#pragma clang loop unroll_count(2)
    for (int kt = 0; kt < D2_ / 32; ++kt) {
        int kb = kt * 32 + hi8;
        union { bf16x16 v; bf16x8 h[2]; } a;
        a.h[0] = *(const bf16x8*)(htbf + arow + kb);
        a.h[1] = *(const bf16x8*)(htbf + arow + kb + 16);
#pragma unroll
        for (int j = 0; j < 4; ++j) {
            bf16x16 bm = *(const bf16x16*)(hWf + (((size_t)(jtb + j) * (D2_ / 32) + kt) * 32 + lane) * 16);
            acc[j] = wmma_bf16(a.v, bm, acc[j]);
        }
    }
    const size_t fbase = ((size_t)b * S_ + (S_ - 1)) * D2_;
#pragma unroll
    for (int j = 0; j < 4; ++j) {
        int n = (jtb + j) * 16 + row;
        float tpv = tp[b * D2_ + n];
        float fin = ht[fbase + n];
#pragma unroll
        for (int r = 0; r < 8; ++r) {
            float ev = tanh_fast(acc[j][r] + tpv);
            atomicAdd(&sLds[r + hi8], ev * fin);
        }
    }
    __syncthreads();
    if (tid < 16) scores[(size_t)b * S_ + s0 + tid] = sLds[tid];
}

// ---------------------------------------------------------------------------
__global__ __launch_bounds__(256)
void k_softmax(const float* __restrict__ scores, float* __restrict__ weight) {
    int b = blockIdx.x, tid = threadIdx.x;
    __shared__ float red[256];
    const float* sc = scores + (size_t)b * S_;
    float mx = -3.0e38f;
    for (int s = tid; s < S_; s += 256) mx = fmaxf(mx, sc[s]);
    red[tid] = mx; __syncthreads();
    for (int o = 128; o > 0; o >>= 1) { if (tid < o) red[tid] = fmaxf(red[tid], red[tid + o]); __syncthreads(); }
    mx = red[0]; __syncthreads();
    float sum = 0.0f;
    for (int s = tid; s < S_; s += 256) sum += __expf(sc[s] - mx);
    red[tid] = sum; __syncthreads();
    for (int o = 128; o > 0; o >>= 1) { if (tid < o) red[tid] += red[tid + o]; __syncthreads(); }
    float inv = 1.0f / red[0];
    for (int s = tid; s < S_; s += 256) weight[(size_t)b * S_ + s] = __expf(sc[s] - mx) * inv;
}

// r[b][d] = sum_s weight[b][s] * ht[b][s][d]   (33 MF; coalesced over d)
__global__ void k_r(const float* __restrict__ weight, const float* __restrict__ ht,
                    float* __restrict__ r) {
    int idx = blockIdx.x * 256 + threadIdx.x;
    if (idx >= B_ * D2_) return;
    int b = idx / D2_, d = idx % D2_;
    const float* w = weight + (size_t)b * S_;
    const float* h = ht + (size_t)b * S_ * D2_ + d;
    float acc = 0.0f;
    for (int s = 0; s < S_; ++s) acc += w[s] * h[(size_t)s * D2_];
    r[idx] = acc;
}

// ---------------------------------------------------------------------------
extern "C" void kernel_launch(void* const* d_in, const int* in_sizes, int n_in,
                              void* d_out, int out_size, void* d_ws, size_t ws_size,
                              hipStream_t stream) {
    (void)in_sizes; (void)n_in; (void)out_size; (void)ws_size;
    const float* tgt  = (const float*)d_in[0];
    const float* x    = (const float*)d_in[1];
    const float* WihF = (const float*)d_in[2];
    const float* WhhF = (const float*)d_in[3];
    const float* bF   = (const float*)d_in[4];
    const float* WihB = (const float*)d_in[5];
    const float* WhhB = (const float*)d_in[6];
    const float* bB   = (const float*)d_in[7];
    const float* tW   = (const float*)d_in[8];
    const float* tb   = (const float*)d_in[9];
    const float* hW   = (const float*)d_in[10];
    float* out = (float*)d_out;

    char* ws = (char*)d_ws;
    size_t off = 0;
    auto alloc = [&](size_t bytes) -> void* {
        void* p = ws + off;
        off = (off + bytes + 255) & ~(size_t)255;
        return p;
    };
    bf16*  xbf  = (bf16*)alloc((size_t)B_ * S_ * EP_ * 2);        // 20 MB
    bf16*  WcF  = (bf16*)alloc((size_t)64 * KT_ * 32 * 16 * 2);   // 1.1 MB
    bf16*  WcB  = (bf16*)alloc((size_t)64 * KT_ * 32 * 16 * 2);   // 1.1 MB
    bf16*  hWf  = (bf16*)alloc((size_t)32 * 16 * 32 * 16 * 2);    // 0.5 MB
    float* ht   = (float*)alloc((size_t)B_ * S_ * D2_ * 4);       // 64 MB
    bf16*  htbf = (bf16*)alloc((size_t)B_ * S_ * D2_ * 2);        // 32 MB
    float* tp   = (float*)alloc((size_t)B_ * D2_ * 4);
    float* sc   = (float*)alloc((size_t)B_ * S_ * 4);

    k_convert_x<<<(B_ * S_ * EP_) / 256, 256, 0, stream>>>(x, xbf);
    k_pack_wc<<<(64 * KT_ * 32 * 16) / 256, 256, 0, stream>>>(WihF, WhhF, WcF);
    k_pack_wc<<<(64 * KT_ * 32 * 16) / 256, 256, 0, stream>>>(WihB, WhhB, WcB);
    k_pack_hw<<<(32 * 16 * 32 * 16) / 256, 256, 0, stream>>>(hW, hWf);
    k_tproj<<<(B_ * D2_) / 256, 256, 0, stream>>>(tgt, tW, tb, tp);

    k_bilstm<<<2, 1024, 0, stream>>>(xbf, WcF, WcB, bF, bB, ht, htbf);

    k_scores<<<(B_ * S_) / 16, 256, 0, stream>>>(htbf, hWf, tp, ht, sc);

    float* weight = out + B_ * D2_;      // output layout: r[16,512] then weight[16,2048]
    k_softmax<<<B_, 256, 0, stream>>>(sc, weight);
    k_r<<<(B_ * D2_) / 256, 256, 0, stream>>>(weight, ht, out);
}
